// GroupedQueryAttention_35820027249327
// MI455X (gfx1250) — matrix operand order repaired and checked by reading, not compiled, not run on hardware
//
#include <hip/hip_runtime.h>

// ---------------------------------------------------------------------------
// Types for CDNA5 WMMA bf16 16x16x32
// ---------------------------------------------------------------------------
typedef __attribute__((ext_vector_type(16))) __bf16         bf16x16;
typedef __attribute__((ext_vector_type(8)))  float          f32x8;
typedef __attribute__((ext_vector_type(8)))  unsigned short u16x8;

// GCC-style vector type matching the async-LDS builtin's parameter
typedef int v4i_g __attribute__((vector_size(16)));
typedef __attribute__((address_space(1))) v4i_g* gptr_b128;
typedef __attribute__((address_space(3))) v4i_g* lptr_b128;

union FragAB {
    u16x8          h[2];
    unsigned short u[16];
    bf16x16        v;
};

__device__ __forceinline__ unsigned short f2bf(float f) {
    unsigned int x = __float_as_uint(f);
    unsigned int r = x + 0x7FFFu + ((x >> 16) & 1u);   // round to nearest even
    return (unsigned short)(r >> 16);
}

#if __has_builtin(__builtin_amdgcn_global_load_async_to_lds_b128)
#define HAVE_ASYNC_LDS 1
#else
#define HAVE_ASYNC_LDS 0
#endif

__device__ __forceinline__ void wait_asynccnt0() {
#if __has_builtin(__builtin_amdgcn_s_wait_asynccnt)
    __builtin_amdgcn_s_wait_asynccnt(0);
#else
    asm volatile("s_wait_asynccnt 0x0" ::: "memory");
#endif
}

// ---------------------------------------------------------------------------
// Kernel 1: fp32 -> bf16 (vectorized x4)
// ---------------------------------------------------------------------------
__global__ void cvt_f32_bf16(const float* __restrict__ in,
                             unsigned short* __restrict__ out, int n4) {
    int i = blockIdx.x * blockDim.x + threadIdx.x;
    if (i < n4) {
        float4 f = reinterpret_cast<const float4*>(in)[i];
        ushort4 o;
        o.x = f2bf(f.x); o.y = f2bf(f.y); o.z = f2bf(f.z); o.w = f2bf(f.w);
        reinterpret_cast<ushort4*>(out)[i] = o;
    }
}

// ---------------------------------------------------------------------------
// Kernel 2: bf16 GEMM  C[M,N] = A[M,K] * B[K,N]
//   block tile 128x128, BK=32, 8 waves (4 along M x 2 along N),
//   wave tile 32x64 -> 8 WMMAs per wave per k-step
// ---------------------------------------------------------------------------
#define BM 128
#define BN 128
#define BK 32
#define AP 48   // LDS pitch (halves); 96 B rows keep 16-B aligned chunks
#define BP 48

__global__ __launch_bounds__(256)
void gemm_bf16(const unsigned short* __restrict__ A,
               const unsigned short* __restrict__ B,
               void* __restrict__ Cout, int M, int N, int K, int outIsF32) {
    __shared__ __align__(16) unsigned short As[BM * AP]; // 12 KB
    __shared__ __align__(16) unsigned short Bt[BN * BP]; // 12 KB (B transposed)

    const int tid  = threadIdx.x;
    const int wid  = tid >> 5;
    const int lane = tid & 31;
    const int ln   = lane & 15;
    const int hi   = lane >> 4;
    const int tileM = blockIdx.y * BM;
    const int tileN = blockIdx.x * BN;
    const int waveM = (wid & 3) * 32;
    const int waveN = (wid >> 2) * 64;

    f32x8 acc[2][4] = {};

    for (int k0 = 0; k0 < K; k0 += BK) {
        // prefetch next k-tiles into GL2 (global_prefetch_b8)
        if (k0 + BK < K) {
            __builtin_prefetch(A + (size_t)(tileM + (tid >> 1)) * K + k0 + BK + (tid & 1) * 16, 0, 0);
            __builtin_prefetch(B + (size_t)(k0 + BK + (tid >> 3)) * N + tileN + (tid & 7) * 16, 0, 0);
        }
        // --- A tile: 128x32 halves; 16B chunks, 2 per thread (async to LDS if available)
        #pragma unroll
        for (int i = 0; i < 2; ++i) {
            int idx = tid + i * 256;
            int row = idx >> 2;
            int kc  = (idx & 3) * 8;
            const unsigned short* gsrc = A + (size_t)(tileM + row) * K + k0 + kc;
            unsigned short*       ldst = &As[row * AP + kc];
#if HAVE_ASYNC_LDS
            __builtin_amdgcn_global_load_async_to_lds_b128(
                (gptr_b128)gsrc, (lptr_b128)ldst, 0, 0);
#else
            *reinterpret_cast<uint4*>(ldst) = *reinterpret_cast<const uint4*>(gsrc);
#endif
        }
        // --- B tile: 32x128, transpose into Bt[n][k]; 2 chunks per thread
        #pragma unroll
        for (int i = 0; i < 2; ++i) {
            int idx = tid + i * 256;
            int kk = idx >> 4;
            int nc = (idx & 15) * 8;
            uint4 d = *reinterpret_cast<const uint4*>(B + (size_t)(k0 + kk) * N + tileN + nc);
            const unsigned short* ds = reinterpret_cast<const unsigned short*>(&d);
            #pragma unroll
            for (int j = 0; j < 8; ++j) Bt[(nc + j) * BP + kk] = ds[j];
        }
#if HAVE_ASYNC_LDS
        wait_asynccnt0();
#endif
        __syncthreads();

        FragAB a[2], bf[4];
        #pragma unroll
        for (int mi = 0; mi < 2; ++mi) {        // A layout: K in {0..7,16..23}+8*hi
            int m  = waveM + mi * 16 + ln;
            int kb = hi * 8;
            a[mi].h[0] = *reinterpret_cast<const u16x8*>(&As[m * AP + kb]);
            a[mi].h[1] = *reinterpret_cast<const u16x8*>(&As[m * AP + kb + 16]);
        }
        #pragma unroll
        for (int ni = 0; ni < 4; ++ni) {        // B layout: K in {0..7,16..23}+8*hi
            int n  = waveN + ni * 16 + ln;
            int kb = hi * 8;
            bf[ni].h[0] = *reinterpret_cast<const u16x8*>(&Bt[n * BP + kb]);
            bf[ni].h[1] = *reinterpret_cast<const u16x8*>(&Bt[n * BP + kb + 16]);
        }
        #pragma unroll
        for (int mi = 0; mi < 2; ++mi)
            #pragma unroll
            for (int ni = 0; ni < 4; ++ni)
                acc[mi][ni] = __builtin_amdgcn_wmma_f32_16x16x32_bf16(
                    false, a[mi].v, false, bf[ni].v, (short)0, acc[mi][ni], false, false);
        __syncthreads();
    }

    // C/D layout: row = r + 8*hi, col = ln
    #pragma unroll
    for (int mi = 0; mi < 2; ++mi)
        #pragma unroll
        for (int ni = 0; ni < 4; ++ni) {
            int col = tileN + waveN + ni * 16 + ln;
            #pragma unroll
            for (int r = 0; r < 8; ++r) {
                int row = tileM + waveM + mi * 16 + r + hi * 8;
                float vv = acc[mi][ni][r];
                if (outIsF32)
                    reinterpret_cast<float*>(Cout)[(size_t)row * N + col] = vv;
                else
                    reinterpret_cast<unsigned short*>(Cout)[(size_t)row * N + col] = f2bf(vv);
            }
        }
}

// ---------------------------------------------------------------------------
// Kernel 3: causal GQA flash attention (bf16 in, bf16 out), D=64
//   grid = B * HQ * (N/128), block = 256 (8 waves), each wave: 16 query rows
// ---------------------------------------------------------------------------
#define NSEQ 2048
#define EDIM 1024
#define HKD  256
#define NHQ  16

__global__ __launch_bounds__(256)
void attn_gqa(const unsigned short* __restrict__ Qb,
              const unsigned short* __restrict__ Kb,
              const unsigned short* __restrict__ Vb,
              unsigned short* __restrict__ Ab) {
    __shared__ __align__(16) unsigned short Plds[8][16 * 32]; // 8 KB, per-wave P scratch

    const int tid  = threadIdx.x;
    const int wid  = tid >> 5;
    const int lane = tid & 31;
    const int ln   = lane & 15;
    const int hi   = lane >> 4;

    const int nQB = NSEQ / 128;               // 16
    int bid = blockIdx.x;
    int qb  = bid % nQB;
    int hq  = (bid / nQB) % NHQ;
    int b   = bid / (nQB * NHQ);
    int hk  = hq >> 2;                        // group size 4
    int qBase = qb * 128 + wid * 16;
    size_t rowOff = (size_t)b * NSEQ;

    // Q fragments: two K-steps over D=64 (A-matrix layout)
    FragAB qf[2];
    {
        const unsigned short* qp = Qb + (rowOff + qBase + ln) * EDIM + hq * 64;
        int kb = hi * 8;
        #pragma unroll
        for (int s = 0; s < 2; ++s) {
            qf[s].h[0] = *reinterpret_cast<const u16x8*>(qp + s * 32 + kb);
            qf[s].h[1] = *reinterpret_cast<const u16x8*>(qp + s * 32 + kb + 16);
        }
    }

    f32x8 o[4] = {};
    float mrow[8], lrow[8];
    #pragma unroll
    for (int r = 0; r < 8; ++r) { mrow[r] = -3.0e38f; lrow[r] = 0.0f; }

    const int lastRow = qBase + 15;
    for (int kBase = 0; kBase <= lastRow; kBase += 32) {
        // ---- S = Q * K^T (16 rows x 32 keys), two 16-col blocks
        f32x8 s[2];
        #pragma unroll
        for (int cb = 0; cb < 2; ++cb) {
            f32x8 accS = {};
            #pragma unroll
            for (int st = 0; st < 2; ++st) {
                FragAB kf;  // B-matrix: B[kdim][key]; K^T gather is contiguous in d
                const unsigned short* kp =
                    Kb + (rowOff + kBase + cb * 16 + ln) * HKD + hk * 64 + st * 32 + hi * 8;
                kf.h[0] = *reinterpret_cast<const u16x8*>(kp);
                kf.h[1] = *reinterpret_cast<const u16x8*>(kp + 16);
                accS = __builtin_amdgcn_wmma_f32_16x16x32_bf16(
                    false, qf[st].v, false, kf.v, (short)0, accS, false, false);
            }
            s[cb] = accS;
        }

        // ---- scale + causal mask + per-row tile max
        float tmax[8];
        #pragma unroll
        for (int r = 0; r < 8; ++r) {
            int row = qBase + r + hi * 8;
            #pragma unroll
            for (int cb = 0; cb < 2; ++cb) {
                int col = kBase + cb * 16 + ln;
                float vv = s[cb][r] * 0.125f;          // 1/sqrt(64)
                if (col > row) vv = -3.0e38f;
                s[cb][r] = vv;
            }
            tmax[r] = fmaxf(s[0][r], s[1][r]);
        }
        #pragma unroll
        for (int off = 1; off < 16; off <<= 1)
            #pragma unroll
            for (int r = 0; r < 8; ++r)
                tmax[r] = fmaxf(tmax[r], __shfl_xor(tmax[r], off, 32));

        // ---- online softmax update
        float alpha[8], tsum[8];
        #pragma unroll
        for (int r = 0; r < 8; ++r) {
            float mnew = fmaxf(mrow[r], tmax[r]);
            alpha[r] = __expf(mrow[r] - mnew);
            mrow[r]  = mnew;
            float p0 = __expf(s[0][r] - mnew);
            float p1 = __expf(s[1][r] - mnew);
            s[0][r] = p0; s[1][r] = p1;
            tsum[r] = p0 + p1;
        }
        #pragma unroll
        for (int off = 1; off < 16; off <<= 1)
            #pragma unroll
            for (int r = 0; r < 8; ++r)
                tsum[r] += __shfl_xor(tsum[r], off, 32);
        #pragma unroll
        for (int r = 0; r < 8; ++r) lrow[r] = lrow[r] * alpha[r] + tsum[r];
        #pragma unroll
        for (int d = 0; d < 4; ++d)
            #pragma unroll
            for (int r = 0; r < 8; ++r) o[d][r] *= alpha[r];

        // ---- P (C-layout) -> LDS -> reload in A-layout (per-wave transpose)
        unsigned short* pw = &Plds[wid][0];
        #pragma unroll
        for (int r = 0; r < 8; ++r) {
            int prow = r + hi * 8;
            pw[prow * 32 + ln]      = f2bf(s[0][r]);
            pw[prow * 32 + 16 + ln] = f2bf(s[1][r]);
        }
        asm volatile("s_wait_dscnt 0x0" ::: "memory"); // wave-local LDS RAW
        FragAB pf;
        {
            const unsigned short* pr = &Plds[wid][ln * 32];
            int kb = hi * 8;
            pf.h[0] = *reinterpret_cast<const u16x8*>(pr + kb);
            pf.h[1] = *reinterpret_cast<const u16x8*>(pr + kb + 16);
        }

        // ---- O += P * V   (four 16-wide d blocks)
        #pragma unroll
        for (int d = 0; d < 4; ++d) {
            FragAB vf;  // B-matrix: B[key][dcol], key = 8*hi + (e&7) + 16*(e>>3)
            #pragma unroll
            for (int e = 0; e < 16; ++e) {
                int kk = kBase + hi * 8 + (e & 7) + (e >> 3) * 16;
                vf.u[e] = Vb[(rowOff + kk) * HKD + hk * 64 + d * 16 + ln];
            }
            o[d] = __builtin_amdgcn_wmma_f32_16x16x32_bf16(
                false, pf.v, false, vf.v, (short)0, o[d], false, false);
        }
    }

    // ---- normalize + store (channel order (h g d) == Q channel order)
    float invl[8];
    #pragma unroll
    for (int r = 0; r < 8; ++r) invl[r] = 1.0f / lrow[r];
    #pragma unroll
    for (int d = 0; d < 4; ++d)
        #pragma unroll
        for (int r = 0; r < 8; ++r) {
            int row = qBase + r + hi * 8;
            Ab[(rowOff + row) * EDIM + hq * 64 + d * 16 + ln] = f2bf(o[d][r] * invl[r]);
        }
}

// ---------------------------------------------------------------------------
// Host launcher
// ---------------------------------------------------------------------------
extern "C" void kernel_launch(void* const* d_in, const int* in_sizes, int n_in,
                              void* d_out, int out_size, void* d_ws, size_t ws_size,
                              hipStream_t stream) {
    (void)in_sizes; (void)n_in; (void)out_size; (void)ws_size;
    const float* x  = (const float*)d_in[0];
    const float* Wq = (const float*)d_in[1];
    const float* Wk = (const float*)d_in[2];
    const float* Wv = (const float*)d_in[3];
    const float* Wo = (const float*)d_in[4];
    float* out = (float*)d_out;

    const int BN_ = 2 * 2048;            // 4096 rows
    unsigned short* xb  = (unsigned short*)d_ws;
    unsigned short* Wqb = xb  + (size_t)BN_ * 1024;
    unsigned short* Wkb = Wqb + (size_t)1024 * 1024;
    unsigned short* Wvb = Wkb + (size_t)1024 * 256;
    unsigned short* Wob = Wvb + (size_t)1024 * 256;
    unsigned short* Qb  = Wob + (size_t)1024 * 1024;
    unsigned short* Kb  = Qb  + (size_t)BN_ * 1024;
    unsigned short* Vb  = Kb  + (size_t)BN_ * 256;
    unsigned short* Ab  = Vb  + (size_t)BN_ * 256;   // total ~33 MB scratch

    auto cvt = [&](const float* src, unsigned short* dst, int n) {
        int n4 = n / 4;
        cvt_f32_bf16<<<(n4 + 255) / 256, 256, 0, stream>>>(src, dst, n4);
    };
    cvt(x,  xb,  BN_ * 1024);
    cvt(Wq, Wqb, 1024 * 1024);
    cvt(Wk, Wkb, 1024 * 256);
    cvt(Wv, Wvb, 1024 * 256);
    cvt(Wo, Wob, 1024 * 1024);

    // projections
    gemm_bf16<<<dim3(1024 / BN, BN_ / BM), 256, 0, stream>>>(xb, Wqb, Qb, BN_, 1024, 1024, 0);
    gemm_bf16<<<dim3(256 / BN,  BN_ / BM), 256, 0, stream>>>(xb, Wkb, Kb, BN_, 256, 1024, 0);
    gemm_bf16<<<dim3(256 / BN,  BN_ / BM), 256, 0, stream>>>(xb, Wvb, Vb, BN_, 256, 1024, 0);

    // attention: B * HQ * (N/128) workgroups
    attn_gqa<<<2 * 16 * (2048 / 128), 256, 0, stream>>>(Qb, Kb, Vb, Ab);

    // output projection -> fp32
    gemm_bf16<<<dim3(1024 / BN, BN_ / BM), 256, 0, stream>>>(Ab, Wob, out, BN_, 1024, 1024, 1);
}
